// LruForgetGate_38860864094330
// MI455X (gfx1250) — matrix operand order repaired and checked by reading, not compiled, not run on hardware
//
#include <hip/hip_runtime.h>
#include <hip/hip_bf16.h>
#include <stddef.h>
#include <stdint.h>

// ---------------------------------------------------------------------------
// LRU forward for MI455X (gfx1250, wave32, WMMA).
//   GEMM1: X(16384x1024) * [nu_w; in_w]^T -> fused sigmoid / lambda / u epilogue
//   Scan : chunked complex linear scan (8 chunks of 512), fp32
//   GEMM2: feature(16384x2048,bf16) * out_w^T -> bias + (b,n,d) transpose
// GEMMs use v_wmma_f32_16x16x32_bf16 (bf16 in, fp32 accumulate) with
// double-buffered LDS tiles staged by GLOBAL_LOAD_ASYNC_TO_LDS_B128
// (ASYNCcnt-tracked DMA overlapping the WMMAs).
// ---------------------------------------------------------------------------

typedef __attribute__((ext_vector_type(16))) __bf16 v16bf;
typedef __attribute__((ext_vector_type(8)))  __bf16 v8bf;
typedef __attribute__((ext_vector_type(8)))  float  v8f;

#define DIMB 4
#define DIMN 4096
#define DIMD 1024
#define DIMM (DIMB * DIMN)      /* 16384 rows (n*4 + b) */
#define N1   (3 * DIMD)         /* GEMM1 output cols: 1024 nu + 2048 in_state */
#define K1   DIMD
#define N2   DIMD
#define K2   (2 * DIMD)
#define NCH  8
#define CLEN (DIMN / NCH)       /* 512 */

#if defined(__HIP_DEVICE_COMPILE__) &&                                  \
    __has_builtin(__builtin_amdgcn_global_load_async_to_lds_b128) &&    \
    __has_builtin(__builtin_amdgcn_s_wait_asynccnt)
#define USE_ASYNC_LDS 1
#else
#define USE_ASYNC_LDS 0
#endif

#if USE_ASYNC_LDS
// Builtin prototype (from hipcc diagnostic): arg0 = int4 AS(1)*, arg1 = int4 AS(3)*.
// Go through integers to form the address-space-qualified pointers: generic
// global pointer == 64-bit address; low 32 bits of a generic LDS pointer ==
// LDS offset (valid 32-bit AS(3) pointer).
typedef int v4i_gcc __attribute__((vector_size(16)));
typedef __attribute__((address_space(1))) v4i_gcc* as1_v4i_p;
typedef __attribute__((address_space(3))) v4i_gcc* as3_v4i_p;

__device__ __forceinline__ as1_v4i_p as_global_v4i(const void* p) {
    return (as1_v4i_p)(unsigned long long)p;
}
__device__ __forceinline__ as3_v4i_p as_lds_v4i(void* p) {
    return (as3_v4i_p)(unsigned int)(unsigned long long)p;
}
#endif

struct GemmArgs {
    const __bf16* A;     // M x K row-major
    const __bf16* B;     // N x K row-major (we compute A * B^T)
    int M, N, K;
    const float* nu_b;   // 1024
    const float* in_b;   // 2048
    const float* out_b;  // 1024
    const float* ct;     // cos(exp(theta_log))
    const float* st;     // sin(exp(theta_log))
    const float* gm;     // exp(gamma_log)
    float* lamr; float* lami; float* ur; float* ui;   // MODE 0 outputs (M x 1024)
    float* out;                                       // MODE 1 output (b,n,d)
};

union FragU { v16bf v; v8bf h[2]; };

// Stage one 128x32 bf16 tile of A and B into LDS (16B granules, 2 per thread).
__device__ __forceinline__ void stage_tile(const __bf16* __restrict__ A,
                                           const __bf16* __restrict__ B,
                                           int K, int m0, int n0, int kk, int tid,
                                           __bf16* lA, __bf16* lB) {
#pragma unroll
    for (int j = 0; j < 2; ++j) {
        const int c   = tid + j * 256;   // 0..511
        const int row = c >> 2;          // 0..127
        const int seg = c & 3;           // 4 segs of 8 bf16
        const __bf16* ga = A + (size_t)(m0 + row) * K + kk + seg * 8;
        const __bf16* gb = B + (size_t)(n0 + row) * K + kk + seg * 8;
        __bf16* la = lA + row * 32 + seg * 8;
        __bf16* lb = lB + row * 32 + seg * 8;
#if USE_ASYNC_LDS
        __builtin_amdgcn_global_load_async_to_lds_b128(as_global_v4i(ga),
                                                       as_lds_v4i(la), 0, 0);
        __builtin_amdgcn_global_load_async_to_lds_b128(as_global_v4i(gb),
                                                       as_lds_v4i(lb), 0, 0);
#else
        *(uint4*)la = *(const uint4*)ga;
        *(uint4*)lb = *(const uint4*)gb;
#endif
    }
}

__device__ __forceinline__ void wait_stage() {
#if USE_ASYNC_LDS
    __builtin_amdgcn_s_wait_asynccnt(0);
#endif
}

// ---------------------------------------------------------------------------
// Tiled WMMA GEMM: 128x128 block tile, K-step 32, double-buffered LDS,
// 8 waves in a 2x4 grid, each wave 64x32 = 4x2 tiles of 16x16.
// ---------------------------------------------------------------------------
template <int MODE>
__global__ __launch_bounds__(256)
void gemm_wmma(GemmArgs g) {
    __shared__ __align__(16) __bf16 lsA[2][128 * 32];
    __shared__ __align__(16) __bf16 lsB[2][128 * 32];

    const int tid  = threadIdx.x;
    const int lane = tid & 31;
    const int w    = tid >> 5;       // wave 0..7
    const int wm   = w >> 2;         // 0..1  (64-row strip)
    const int wn   = w & 3;          // 0..3  (32-col strip)
    const int m0   = blockIdx.y * 128;
    const int n0   = blockIdx.x * 128;

    const int mrow  = lane & 15;
    const int khalf = lane >> 4;

    v8f acc[4][2];
    const v8f vzero = {0.f, 0.f, 0.f, 0.f, 0.f, 0.f, 0.f, 0.f};
#pragma unroll
    for (int mi = 0; mi < 4; ++mi)
#pragma unroll
        for (int ni = 0; ni < 2; ++ni) acc[mi][ni] = vzero;

    // Prologue: stage tile 0 into buffer 0.
    stage_tile(g.A, g.B, g.K, m0, n0, 0, tid, lsA[0], lsB[0]);
    wait_stage();
    __syncthreads();

    for (int kk = 0; kk < g.K; kk += 32) {
        const int cur = (kk >> 5) & 1;
        // Kick off DMA for the next K-slice into the other buffer; it runs
        // concurrently with the WMMA work below (ASYNCcnt-tracked).
        if (kk + 32 < g.K)
            stage_tile(g.A, g.B, g.K, m0, n0, kk + 32, tid,
                       lsA[cur ^ 1], lsB[cur ^ 1]);

        // A fragments: 16x32, lane holds row (lane&15); lanes<16: K 0-7 & 16-23,
        // lanes>=16: K 8-15 & 24-31 (ISA 7.12.2 bf16 A layout).
        FragU af[4];
#pragma unroll
        for (int mi = 0; mi < 4; ++mi) {
            const __bf16* p = lsA[cur] + (wm * 64 + mi * 16 + mrow) * 32 + khalf * 8;
            af[mi].h[0] = *(const v8bf*)p;
            af[mi].h[1] = *(const v8bf*)(p + 16);
        }
        // B fragments: 32x16, lane holds col (lane&15); lanes<16: K 0-7 & 16-23,
        // lanes>=16: K 8-15 & 24-31 (same K order as A).
        FragU bfr[2];
#pragma unroll
        for (int ni = 0; ni < 2; ++ni) {
            const __bf16* p = lsB[cur] + (wn * 32 + ni * 16 + mrow) * 32 + khalf * 8;
            bfr[ni].h[0] = *(const v8bf*)p;
            bfr[ni].h[1] = *(const v8bf*)(p + 16);
        }
#pragma unroll
        for (int mi = 0; mi < 4; ++mi)
#pragma unroll
            for (int ni = 0; ni < 2; ++ni)
                acc[mi][ni] = __builtin_amdgcn_wmma_f32_16x16x32_bf16(
                    false, af[mi].v, false, bfr[ni].v,
                    (short)0, acc[mi][ni], false, false);

        // Next tile's DMA must have landed (and everyone must be done reading
        // the buffer we'll overwrite next round) before we flip buffers.
        wait_stage();
        __syncthreads();
    }

    // Epilogue. C/D layout: lane l, vgpr v -> M = v + (l>=16)*8, N = l&15.
    const int mloc = (lane >> 4) * 8;
    const int nloc = lane & 15;
#pragma unroll
    for (int mi = 0; mi < 4; ++mi) {
#pragma unroll
        for (int ni = 0; ni < 2; ++ni) {
#pragma unroll
            for (int vv = 0; vv < 8; ++vv) {
                const int m = m0 + wm * 64 + mi * 16 + mloc + vv;
                const int n = n0 + wn * 32 + ni * 16 + nloc;
                const float v = acc[mi][ni][vv];
                if (MODE == 0) {
                    if (n < DIMD) {                 // nu branch
                        const float z  = v + g.nu_b[n];
                        const float nu = 1.0f / (1.0f + __expf(-z));
                        g.lamr[(size_t)m * DIMD + n] = nu * g.ct[n];
                        g.lami[(size_t)m * DIMD + n] = nu * g.st[n];
                    } else {                        // input_state branch
                        const int   e = n - DIMD;
                        const float z = v + g.in_b[e];
                        const int   d = e >> 1;
                        const float u = g.gm[d] * z;
                        if (e & 1) g.ui[(size_t)m * DIMD + d] = u;
                        else       g.ur[(size_t)m * DIMD + d] = u;
                    }
                } else {                            // output projection
                    const float z   = v + g.out_b[n];
                    const int   bi  = m & 3;
                    const int   seq = m >> 2;
                    g.out[((size_t)bi * DIMN + seq) * DIMD + n] = z;
                }
            }
        }
    }
}

// ---------------------------------------------------------------------------
// Param prep + bf16 converts
// ---------------------------------------------------------------------------
__global__ void prep_params(const float* __restrict__ theta_log,
                            const float* __restrict__ gamma_log,
                            float* ct, float* st, float* gm) {
    const int i = blockIdx.x * blockDim.x + threadIdx.x;
    if (i < DIMD) {
        const float th = __expf(theta_log[i]);
        ct[i] = __cosf(th);
        st[i] = __sinf(th);
        gm[i] = __expf(gamma_log[i]);
    }
}

// x (b, n, d) fp32 -> X (n*4+b, d) bf16
__global__ __launch_bounds__(256)
void convert_x(const float* __restrict__ x, __bf16* __restrict__ xb) {
    const size_t gid = (size_t)blockIdx.x * blockDim.x + threadIdx.x;
    const size_t o   = gid * 4;
    const size_t m   = o >> 10;
    const int    d   = (int)(o & 1023);
    const int    ni  = (int)(m >> 2);
    const int    bi  = (int)(m & 3);
    const float4 v = *(const float4*)(x + ((size_t)bi * DIMN + ni) * DIMD + d);
    xb[o + 0] = (__bf16)v.x;
    xb[o + 1] = (__bf16)v.y;
    xb[o + 2] = (__bf16)v.z;
    xb[o + 3] = (__bf16)v.w;
}

// [nu_w (1024x1024); in_w (2048x1024)] fp32 -> bf16
__global__ __launch_bounds__(256)
void convert_wcat(const float* __restrict__ nu_w, const float* __restrict__ in_w,
                  __bf16* __restrict__ wc) {
    const size_t o = (size_t)blockIdx.x * blockDim.x + threadIdx.x;
    const float  s = (o < (size_t)DIMD * DIMD) ? nu_w[o] : in_w[o - (size_t)DIMD * DIMD];
    wc[o] = (__bf16)s;
}

// out_w (1024x2048) fp32 -> bf16
__global__ __launch_bounds__(256)
void convert_outw(const float* __restrict__ out_w, __bf16* __restrict__ ow) {
    const size_t o = (size_t)blockIdx.x * blockDim.x + threadIdx.x;
    ow[o] = (__bf16)out_w[o];
}

// ---------------------------------------------------------------------------
// Chunked complex linear scan: h_t = lam_t * h_{t-1} + u_t
// ---------------------------------------------------------------------------
__global__ __launch_bounds__(256)
void scan_phase1(const float* __restrict__ lamr, const float* __restrict__ lami,
                 const float* __restrict__ ur,   const float* __restrict__ ui,
                 float* cpr, float* cpi, float* chr, float* chi) {
    const int d  = blockIdx.x * 256 + threadIdx.x;  // 0..1023
    const int bi = blockIdx.y;                      // 0..3
    const int ch = blockIdx.z;                      // 0..7
    float pr = 1.f, pi = 0.f, hr = 0.f, hi = 0.f;
    for (int t = 0; t < CLEN; ++t) {
        const int    ni  = ch * CLEN + t;
        const size_t idx = ((size_t)(ni * 4 + bi) << 10) + d;
        const float lr = lamr[idx], li = lami[idx];
        const float vr = ur[idx],   vi = ui[idx];
        const float nhr = lr * hr - li * hi + vr;
        const float nhi = lr * hi + li * hr + vi;
        hr = nhr; hi = nhi;
        const float npr = lr * pr - li * pi;
        const float npi = lr * pi + li * pr;
        pr = npr; pi = npi;
    }
    const size_t c = ((size_t)ch * 4 + bi) * DIMD + d;
    cpr[c] = pr; cpi[c] = pi; chr[c] = hr; chi[c] = hi;
}

__global__ void scan_phase2(const float* __restrict__ cpr, const float* __restrict__ cpi,
                            const float* __restrict__ chr, const float* __restrict__ chi,
                            float* sr, float* si) {
    const int c  = blockIdx.x * 256 + threadIdx.x;  // channel 0..4095
    const int bi = c >> 10;
    const int d  = c & 1023;
    float hr = 0.f, hi = 0.f;
    for (int ch = 0; ch < NCH; ++ch) {
        const size_t idx = ((size_t)ch * 4 + bi) * DIMD + d;
        sr[idx] = hr; si[idx] = hi;                 // carry at chunk start
        const float pr = cpr[idx], pi = cpi[idx];
        const float nhr = pr * hr - pi * hi + chr[idx];
        const float nhi = pr * hi + pi * hr + chi[idx];
        hr = nhr; hi = nhi;
    }
}

__global__ __launch_bounds__(256)
void scan_phase3(const float* __restrict__ lamr, const float* __restrict__ lami,
                 const float* __restrict__ ur,   const float* __restrict__ ui,
                 const float* __restrict__ sr,   const float* __restrict__ si,
                 __bf16* __restrict__ feat) {
    const int d  = blockIdx.x * 256 + threadIdx.x;
    const int bi = blockIdx.y;
    const int ch = blockIdx.z;
    const size_t cidx = ((size_t)ch * 4 + bi) * DIMD + d;
    float hr = sr[cidx], hi = si[cidx];
    for (int t = 0; t < CLEN; ++t) {
        const int    ni  = ch * CLEN + t;
        const size_t m   = (size_t)(ni * 4 + bi);
        const size_t idx = (m << 10) + d;
        const float lr = lamr[idx], li = lami[idx];
        const float vr = ur[idx],   vi = ui[idx];
        const float nhr = lr * hr - li * hi + vr;
        const float nhi = lr * hi + li * hr + vi;
        hr = nhr; hi = nhi;
        feat[m * 2048 + d]        = (__bf16)hr;
        feat[m * 2048 + 1024 + d] = (__bf16)hi;
    }
}

// ---------------------------------------------------------------------------
// Launch
// ---------------------------------------------------------------------------
extern "C" void kernel_launch(void* const* d_in, const int* in_sizes, int n_in,
                              void* d_out, int out_size, void* d_ws, size_t ws_size,
                              hipStream_t stream) {
    const float* x         = (const float*)d_in[0];
    const float* theta_log = (const float*)d_in[1];
    const float* gamma_log = (const float*)d_in[2];
    const float* nu_w      = (const float*)d_in[3];
    const float* nu_b      = (const float*)d_in[4];
    const float* in_w      = (const float*)d_in[5];
    const float* in_b      = (const float*)d_in[6];
    const float* out_w     = (const float*)d_in[7];
    const float* out_b     = (const float*)d_in[8];
    float* out = (float*)d_out;

    char*  ws  = (char*)d_ws;
    size_t off = 0;
    auto alloc = [&](size_t bytes) -> void* {
        void* p = ws + off;
        off = (off + bytes + 255) & ~(size_t)255;
        return p;
    };
    __bf16* XB   = (__bf16*)alloc((size_t)DIMM * K1 * 2);
    __bf16* WC   = (__bf16*)alloc((size_t)N1 * K1 * 2);
    __bf16* OW   = (__bf16*)alloc((size_t)N2 * K2 * 2);
    __bf16* FEAT = (__bf16*)alloc((size_t)DIMM * K2 * 2);
    float*  CT   = (float*)alloc(DIMD * 4);
    float*  ST   = (float*)alloc(DIMD * 4);
    float*  GM   = (float*)alloc(DIMD * 4);
    float*  LAMR = (float*)alloc((size_t)DIMM * DIMD * 4);
    float*  LAMI = (float*)alloc((size_t)DIMM * DIMD * 4);
    float*  UR   = (float*)alloc((size_t)DIMM * DIMD * 4);
    float*  UI   = (float*)alloc((size_t)DIMM * DIMD * 4);
    float*  CPR  = (float*)alloc((size_t)NCH * DIMB * DIMD * 4);
    float*  CPI  = (float*)alloc((size_t)NCH * DIMB * DIMD * 4);
    float*  CHR  = (float*)alloc((size_t)NCH * DIMB * DIMD * 4);
    float*  CHI  = (float*)alloc((size_t)NCH * DIMB * DIMD * 4);
    float*  SR   = (float*)alloc((size_t)NCH * DIMB * DIMD * 4);
    float*  SI   = (float*)alloc((size_t)NCH * DIMB * DIMD * 4);

    prep_params<<<4, 256, 0, stream>>>(theta_log, gamma_log, CT, ST, GM);
    convert_x<<<(DIMM * K1 / 4) / 256, 256, 0, stream>>>(x, XB);
    convert_wcat<<<(N1 * K1) / 256, 256, 0, stream>>>(nu_w, in_w, WC);
    convert_outw<<<(N2 * K2) / 256, 256, 0, stream>>>(out_w, OW);

    GemmArgs g1{};
    g1.A = XB; g1.B = WC; g1.M = DIMM; g1.N = N1; g1.K = K1;
    g1.nu_b = nu_b; g1.in_b = in_b; g1.out_b = out_b;
    g1.ct = CT; g1.st = ST; g1.gm = GM;
    g1.lamr = LAMR; g1.lami = LAMI; g1.ur = UR; g1.ui = UI; g1.out = out;
    gemm_wmma<0><<<dim3(N1 / 128, DIMM / 128), 256, 0, stream>>>(g1);

    scan_phase1<<<dim3(DIMD / 256, DIMB, NCH), 256, 0, stream>>>(
        LAMR, LAMI, UR, UI, CPR, CPI, CHR, CHI);
    scan_phase2<<<(DIMB * DIMD) / 256, 256, 0, stream>>>(CPR, CPI, CHR, CHI, SR, SI);
    scan_phase3<<<dim3(DIMD / 256, DIMB, NCH), 256, 0, stream>>>(
        LAMR, LAMI, UR, UI, SR, SI, FEAT);

    GemmArgs g2{};
    g2.A = FEAT; g2.B = OW; g2.M = DIMM; g2.N = N2; g2.K = K2;
    g2.nu_b = nu_b; g2.in_b = in_b; g2.out_b = out_b;
    g2.ct = CT; g2.st = ST; g2.gm = GM;
    g2.lamr = LAMR; g2.lami = LAMI; g2.ur = UR; g2.ui = UI; g2.out = out;
    gemm_wmma<1><<<dim3(N2 / 128, DIMM / 128), 256, 0, stream>>>(g2);
}
